// VectorQuantizer_3015067041859
// MI455X (gfx1250) — matrix operand order repaired and checked by reading, not compiled, not run on hardware
//
#include <hip/hip_runtime.h>

// ---------------------------------------------------------------------------
// VQ-VAE vector quantizer for MI455X (gfx1250, wave32, WMMA + TDM).
// N rows (262144) x D=128 fp32 inputs, K=1024 codes.
// Distance matmul in bf16 WMMA (16x16x32), argmin = |e|^2 - 2 x.e  (drop |x|^2).
// Codebook slabs DMA'd into LDS by the Tensor Data Mover (double-buffered),
// with TDM pad_enable providing the 256B->272B row padding (bank-conflict-free
// ds_load_b128 B-fragment reads).
// ---------------------------------------------------------------------------

typedef __attribute__((ext_vector_type(16))) __bf16 v16bf;
typedef __attribute__((ext_vector_type(8)))  float  v8f;
typedef __attribute__((ext_vector_type(4)))  unsigned int u32x4;
typedef __attribute__((ext_vector_type(8)))  int  i32x8;
typedef __attribute__((ext_vector_type(4)))  int  i32x4;

union Frag16 {
    unsigned short u[16];
    uint4          q[2];
    v16bf          v;
};

__device__ __forceinline__ unsigned short f2bf(float f) {
    unsigned int u = __float_as_uint(f);
    u += 0x7FFFu + ((u >> 16) & 1u);   // round-to-nearest-even
    return (unsigned short)(u >> 16);
}

// ---------------------------------------------------------------------------
// Kernel A: codebook prep.  One wave per code: bf16 copy + |e|^2.
// ---------------------------------------------------------------------------
__global__ void vq_prep(const float* __restrict__ E,
                        unsigned short* __restrict__ Ebf,
                        float* __restrict__ normE) {
    const int code = blockIdx.x;
    const int lane = threadIdx.x;          // 32 threads
    const float4 e = *(const float4*)(E + (size_t)code * 128 + lane * 4);
    unsigned int* dst = (unsigned int*)(Ebf + (size_t)code * 128);
    dst[lane * 2 + 0] = (unsigned)f2bf(e.x) | ((unsigned)f2bf(e.y) << 16);
    dst[lane * 2 + 1] = (unsigned)f2bf(e.z) | ((unsigned)f2bf(e.w) << 16);
    float s = e.x * e.x + e.y * e.y + e.z * e.z + e.w * e.w;
    for (int off = 16; off; off >>= 1) s += __shfl_xor(s, off, 32);
    if (lane == 0) normE[code] = s;
}

// ---------------------------------------------------------------------------
// Kernel B: WMMA distance + argmin, TDM-fed double-buffered LDS slabs.
// WG = 256 threads (8 waves). Each wave owns 16 rows (A tile in registers).
// ---------------------------------------------------------------------------
#define SLAB_CODES 64
#define LDS_ROW    136   // ushorts per code row: 128 dims + 8 pad (272 B)
#define SLAB_U16   (SLAB_CODES * LDS_ROW)        // 8704 ushorts = 17408 B

// TDM: async-copy one slab (64 codes x 256B rows) global->LDS, padding each
// 64-DWORD row with 4 DWORDs (256B -> 272B LDS row stride).
__device__ __forceinline__ void tdm_load_slab(const unsigned short* srcGlobal,
                                              unsigned ldsByteAddr) {
    const unsigned long long ga = (unsigned long long)(uintptr_t)srcGlobal;
    u32x4 g0;
    g0[0] = 1u;                                    // count=1 (valid user D#)
    g0[1] = ldsByteAddr;                           // lds_addr
    g0[2] = (unsigned)(ga & 0xFFFFFFFFu);          // global_addr[31:0]
    g0[3] = (unsigned)((ga >> 32) & 0x1FFFFFFu)    // global_addr[56:32]
            | (2u << 30);                          // type=2 ("image")
    i32x8 g1;
    g1[0] = (2 << 16)     // data_size = 4 bytes
          | (1 << 20)     // pad_enable
          | (5 << 22)     // pad_interval: every 64 DWORDs
          | (3 << 25);    // pad_amount: 4 DWORDs (16B)
    g1[1] = (64 << 16);   // tensor_dim0[15:0] = 64 DWORDs/row
    g1[2] = (64 << 16);   // tensor_dim0 hi=0 | tensor_dim1[15:0] = 64 rows
    g1[3] = (64 << 16);   // tensor_dim1 hi=0 | tile_dim0 = 64
    g1[4] = 64;           // tile_dim1 = 64 rows, tile_dim2 = 0
    g1[5] = 64;           // tensor_dim0_stride = 64 DWORDs
    g1[6] = 0;
    g1[7] = 0;
    const i32x4 z4 = {0, 0, 0, 0};                 // 2D tensor: groups 2/3 unused
    const i32x8 z8 = {0, 0, 0, 0, 0, 0, 0, 0};     // extra operand (6-arg form)
    __builtin_amdgcn_tensor_load_to_lds(g0, g1, z4, z4, z8, 0);
}

__global__ void __launch_bounds__(256)
vq_argmin(const float* __restrict__ X,
          const unsigned short* __restrict__ Ebf,
          const float* __restrict__ normE,
          unsigned int* __restrict__ outIdx,
          int K) {
    __shared__ unsigned short eSlab[2 * SLAB_U16];   // double buffer, 34816 B

    const int  tid  = threadIdx.x;
    const int  wave = tid >> 5;
    const int  lane = tid & 31;
    const int  h    = lane >> 4;           // lane half (A/B K-split, C row-split)
    const int  m    = lane & 15;
    const long rowBase = (long)blockIdx.x * 128 + wave * 16;
    const long row     = rowBase + m;

    const unsigned ldsBase = (unsigned)(uintptr_t)&eSlab[0];  // LDS byte addr

    // --- A fragments: 16 rows x 128 dims as 4 chunks of K=32 (bf16) --------
    // 16-bit A 16x32 layout: lane half h holds dims {8h..8h+7, 16+8h..16+8h+7}.
    Frag16 a[4];
    const float* xr = X + row * 128;
#pragma unroll
    for (int c = 0; c < 4; ++c) {
        const float* p = xr + 32 * c + 8 * h;
        float4 f0 = *(const float4*)(p);
        float4 f1 = *(const float4*)(p + 4);
        float4 f2 = *(const float4*)(p + 16);
        float4 f3 = *(const float4*)(p + 20);
        a[c].u[0]  = f2bf(f0.x); a[c].u[1]  = f2bf(f0.y);
        a[c].u[2]  = f2bf(f0.z); a[c].u[3]  = f2bf(f0.w);
        a[c].u[4]  = f2bf(f1.x); a[c].u[5]  = f2bf(f1.y);
        a[c].u[6]  = f2bf(f1.z); a[c].u[7]  = f2bf(f1.w);
        a[c].u[8]  = f2bf(f2.x); a[c].u[9]  = f2bf(f2.y);
        a[c].u[10] = f2bf(f2.z); a[c].u[11] = f2bf(f2.w);
        a[c].u[12] = f2bf(f3.x); a[c].u[13] = f2bf(f3.y);
        a[c].u[14] = f2bf(f3.z); a[c].u[15] = f2bf(f3.w);
    }

    float        bestVal[8];
    unsigned int bestIdx[8];
#pragma unroll
    for (int i = 0; i < 8; ++i) { bestVal[i] = 3.4e38f; bestIdx[i] = 0u; }

    const int nSlabs = K / SLAB_CODES;     // 16

    // Prologue: wave 0 DMAs slab 0 into buffer 0.
    if (tid < 32) {
        tdm_load_slab(Ebf, ldsBase);
        __builtin_amdgcn_s_wait_tensorcnt(0);
    }
    __syncthreads();

    for (int slab = 0; slab < nSlabs; ++slab) {
        const int buf = slab & 1;
        // Kick off next slab's DMA into the other buffer (overlaps compute).
        if (slab + 1 < nSlabs && tid < 32)
            tdm_load_slab(Ebf + (size_t)(slab + 1) * SLAB_CODES * 128,
                          ldsBase + (unsigned)(buf ^ 1) * (SLAB_U16 * 2u));
        const unsigned bufOff = (unsigned)buf * SLAB_U16;   // in ushorts

#pragma unroll
        for (int tile = 0; tile < SLAB_CODES / 16; ++tile) {
            const int codeN = tile * 16 + m;   // this lane's B column (code)
            v8f acc = {};
#pragma unroll
            for (int c = 0; c < 4; ++c) {
                // 16-bit B 32x16: lane half h holds K = {8h..8h+7, 16+8h..16+8h+7} of column m.
                Frag16 b;
                const uint4* lp = (const uint4*)
                    &eSlab[bufOff + codeN * LDS_ROW + c * 32 + h * 8];
                b.q[0] = lp[0];
                b.q[1] = lp[2];
                acc = __builtin_amdgcn_wmma_f32_16x16x32_bf16(
                    false, a[c].v, false, b.v, (short)0, acc, false, false);
            }
            const float        nrm = normE[slab * SLAB_CODES + tile * 16 + m];
            const unsigned int cg  = (unsigned)(slab * SLAB_CODES + tile * 16 + m);
#pragma unroll
            for (int i = 0; i < 8; ++i) {
                const float s = nrm - 2.0f * acc[i];   // |e|^2 - 2 x.e
                if (s < bestVal[i]) { bestVal[i] = s; bestIdx[i] = cg; }
            }
        }

        // Publish next buffer: issuing wave drains TENSORcnt, then barrier.
        if (tid < 32) __builtin_amdgcn_s_wait_tensorcnt(0);
        __syncthreads();
    }

    // Cross-lane argmin within each 16-lane group (C: row i+8h, col = lane&15).
#pragma unroll
    for (int off = 8; off; off >>= 1) {
#pragma unroll
        for (int i = 0; i < 8; ++i) {
            const float        ov = __shfl_xor(bestVal[i], off, 32);
            const unsigned int oi = __shfl_xor(bestIdx[i], off, 32);
            if (ov < bestVal[i] || (ov == bestVal[i] && oi < bestIdx[i])) {
                bestVal[i] = ov; bestIdx[i] = oi;
            }
        }
    }
    if (m == 0) {
#pragma unroll
        for (int i = 0; i < 8; ++i)
            outIdx[rowBase + h * 8 + i] = bestIdx[i];
    }
}

// ---------------------------------------------------------------------------
// Kernel C: gather quantized rows + deterministic per-block MSE partial.
// One wave per row; lane covers 4 dims.
// ---------------------------------------------------------------------------
__global__ void __launch_bounds__(256)
vq_gather(const float* __restrict__ X, const float* __restrict__ E,
          const unsigned int* __restrict__ idx,
          float* __restrict__ outQ, float* __restrict__ partials) {
    __shared__ float wsum[8];
    const int  tid  = threadIdx.x;
    const int  wave = tid >> 5;
    const int  lane = tid & 31;
    const long row  = (long)blockIdx.x * 8 + wave;

    const unsigned int k = idx[row];
    const float4 q = *(const float4*)(E + (size_t)k * 128 + lane * 4);
    const float4 x = *(const float4*)(X + row * 128 + lane * 4);
    float* o = outQ + row * 128 + lane * 4;   // outQ = d_out+1 -> 4B aligned only
    o[0] = q.x; o[1] = q.y; o[2] = q.z; o[3] = q.w;

    const float dx = q.x - x.x, dy = q.y - x.y, dz = q.z - x.z, dw = q.w - x.w;
    float s = dx * dx + dy * dy + dz * dz + dw * dw;
    for (int off = 16; off; off >>= 1) s += __shfl_xor(s, off, 32);
    if (lane == 0) wsum[wave] = s;
    __syncthreads();
    if (tid == 0) {
        float t = 0.f;
#pragma unroll
        for (int i = 0; i < 8; ++i) t += wsum[i];
        partials[blockIdx.x] = t;
    }
}

// ---------------------------------------------------------------------------
// Kernel D: deterministic final reduction -> loss = 1.25 * mean((q-x)^2).
// ---------------------------------------------------------------------------
__global__ void vq_loss(const float* __restrict__ partials, int n,
                        float* __restrict__ out, float invNumel) {
    __shared__ float red[256];
    float s = 0.f;
    for (int i = threadIdx.x; i < n; i += 256) s += partials[i];
    red[threadIdx.x] = s;
    __syncthreads();
    for (int o = 128; o; o >>= 1) {
        if ((int)threadIdx.x < o) red[threadIdx.x] += red[threadIdx.x + o];
        __syncthreads();
    }
    if (threadIdx.x == 0) out[0] = 1.25f * red[0] * invNumel;
}

// ---------------------------------------------------------------------------
extern "C" void kernel_launch(void* const* d_in, const int* in_sizes, int n_in,
                              void* d_out, int out_size, void* d_ws, size_t ws_size,
                              hipStream_t stream) {
    const float* X = (const float*)d_in[0];        // [64,4096,128] fp32
    const float* E = (const float*)d_in[1];        // [1024,128] fp32
    const long nX = (long)in_sizes[0];             // N*128
    const int  K  = in_sizes[1] / 128;             // 1024
    const long N  = nX / 128;                      // 262144
    float* out = (float*)d_out;                    // [loss, quantized_st...]

    // Workspace layout (all 16B aligned): bf16 codebook, norms, indices, partials.
    char*           ws       = (char*)d_ws;
    unsigned short* Ebf      = (unsigned short*)ws;                       // K*128 u16
    float*          normE    = (float*)(ws + (size_t)K * 128 * 2);        // K f32
    unsigned int*   idx      = (unsigned int*)((char*)normE + (size_t)K * 4);  // N u32
    float*          partials = (float*)((char*)idx + (size_t)N * 4);      // N/8 f32
    const int nBlocksC = (int)(N / 8);

    vq_prep  <<<K,              32,  0, stream>>>(E, Ebf, normE);
    vq_argmin<<<(int)(N / 128), 256, 0, stream>>>(X, Ebf, normE, idx, K);
    vq_gather<<<nBlocksC,       256, 0, stream>>>(X, E, idx, out + 1, partials);
    vq_loss  <<<1,              256, 0, stream>>>(partials, nBlocksC, out,
                                                  1.0f / (float)nX);
}
